// ValueNetwork_23785528886024
// MI455X (gfx1250) — compile-verified, matrix operand order checked by reading, not run on hardware
//
#include <hip/hip_runtime.h>
#include <cstddef>
#include <cstdint>

// ---------------------------------------------------------------------------
// Problem dims (fixed by the reference)
// ---------------------------------------------------------------------------
#define S_LEN 256
#define B_SZ  256
#define D_IN  512
#define H_SZ  512
#define G3H   1536   // 3*H
#define M_SZ  512
#define SB    (S_LEN * B_SZ)   // 65536 rows of the big Gi GEMM

typedef __bf16 bf16_t;
typedef __attribute__((ext_vector_type(16))) __bf16 v16bf;
typedef __attribute__((ext_vector_type(8)))  float  v8f;

// LDS row strides (bytes), padded +16B so lane r hits bank 4*r (no conflicts)
#define LDS_B_STRIDE 272     // 128 bf16 (256B) + 16B pad   (gemm_gi chunk)
#define LDS_A_STRIDE 1040    // 512 bf16 (1024B) + 16B pad  (gru_step h rows)

__device__ inline unsigned short f32_to_bf16_bits(float f) {
  unsigned int u = __float_as_uint(f);
  u += 0x7fffu + ((u >> 16) & 1u);   // round-to-nearest-even
  return (unsigned short)(u >> 16);
}

__device__ inline v8f v8f_zero() {
  v8f z = {0.f, 0.f, 0.f, 0.f, 0.f, 0.f, 0.f, 0.f};
  return z;
}

// CDNA5 16-bit A/B fragment: lane l (r=l&15, hi=l>>4) holds row/col r,
// K = [k0+hi*8, +8) and [k0+16+hi*8, +8)  -> two 16B contiguous loads.
__device__ inline v16bf load_frag(const bf16_t* __restrict__ row, int k0, int hi) {
  union { uint4 u[2]; v16bf v; } t;
  t.u[0] = *(const uint4*)(row + k0 + hi * 8);
  t.u[1] = *(const uint4*)(row + k0 + 16 + hi * 8);
  return t.v;
}

// Same fragment, sourced from an LDS-staged tile (row-major, padded stride).
__device__ inline v16bf load_frag_lds(const char* base, int rowStride,
                                      int row, int k0, int hi) {
  const char* p = base + (size_t)row * rowStride + k0 * 2 + hi * 16;
  union { uint4 u[2]; v16bf v; } t;
  t.u[0] = *(const uint4*)(p);
  t.u[1] = *(const uint4*)(p + 32);
  return t.v;
}

__device__ inline float sigmoidf_(float x) { return 1.0f / (1.0f + __expf(-x)); }

// ---------------------------------------------------------------------------
// Utility kernels: f32 -> bf16 convert, zero fill
// ---------------------------------------------------------------------------
__global__ void cvt_bf16_kernel(const float* __restrict__ src,
                                unsigned short* __restrict__ dst, size_t n) {
  size_t stride = (size_t)gridDim.x * blockDim.x;
  for (size_t i = (size_t)blockIdx.x * blockDim.x + threadIdx.x; i < n; i += stride)
    dst[i] = f32_to_bf16_bits(src[i]);
}

__global__ void zero_kernel(unsigned int* __restrict__ p, size_t nwords) {
  size_t stride = (size_t)gridDim.x * blockDim.x;
  for (size_t i = (size_t)blockIdx.x * blockDim.x + threadIdx.x; i < nwords; i += stride)
    p[i] = 0u;
}

// ---------------------------------------------------------------------------
// Phase B: Gi[s*B+b, n] = sum_k input[s,b,k] * w_ih[n,k]
// Block = 128 rows x 128 cols (8 waves, each 16 rows x 128 cols, 8 accums).
// Weight (B operand) chunk staged in LDS per 128-wide K-chunk; the inner
// (k0, nt) loop is flattened and software-pipelined TWO fragments deep so two
// ds_load pairs stay in flight behind every WMMA (steady-state partial waits).
// Dynamic LDS: 128 * LDS_B_STRIDE = 34816 B.
// ---------------------------------------------------------------------------
__global__ void gemm_gi_kernel(const unsigned short* __restrict__ Xb,   // [SB,512] bf16
                               const unsigned short* __restrict__ Wb,   // [1536,512] bf16
                               float* __restrict__ Gi) {                // [SB,1536] f32
  extern __shared__ char smem[];
  const int tid  = threadIdx.x;
  const int wave = tid >> 5;
  const int lane = tid & 31;
  const int r    = lane & 15;
  const int hi   = lane >> 4;
  const int m0   = blockIdx.x * 128 + wave * 16;
  const int n0   = blockIdx.y * 128;

  const bf16_t* X = (const bf16_t*)Xb;
  const bf16_t* W = (const bf16_t*)Wb;
  const bf16_t* arow = X + (size_t)(m0 + r) * D_IN;
  __builtin_prefetch(arow, 0, 1);   // global_prefetch_b8 on the A row stream

  v8f acc[8];
#pragma unroll
  for (int i = 0; i < 8; ++i) acc[i] = v8f_zero();

  // Cooperative staging map: 2 threads per weight row, 128B each.
  const int srow  = tid >> 1;
  const int shalf = tid & 1;

  for (int kc = 0; kc < D_IN; kc += 128) {
    __syncthreads();   // previous chunk fully consumed
    {
      const uint4* src = (const uint4*)(W + (size_t)(n0 + srow) * D_IN + kc + shalf * 64);
      uint4* dst = (uint4*)(smem + (size_t)srow * LDS_B_STRIDE + shalf * 128);
#pragma unroll
      for (int i = 0; i < 8; ++i) dst[i] = src[i];
    }
    __syncthreads();   // chunk visible to all waves

    // Flattened, fully unrolled, 2-deep pipelined: it = k-step*8 + nt.
    v16bf a  = load_frag(arow, kc, hi);
    v16bf an = a;
    v16bf b0 = load_frag_lds(smem, LDS_B_STRIDE, r, 0, hi);
    v16bf b1 = load_frag_lds(smem, LDS_B_STRIDE, 16 + r, 0, hi);
#pragma unroll
    for (int it = 0; it < 32; ++it) {
      const int nt  = it & 7;
      const int it2 = it + 2;
      v16bf b2 = b1;
      if (it2 < 32)
        b2 = load_frag_lds(smem, LDS_B_STRIDE, (it2 & 7) * 16 + r,
                           (it2 >> 3) * 32, hi);
      if (nt == 6 && it2 < 32)   // next k-step's A, two iterations early
        an = load_frag(arow, kc + (it2 >> 3) * 32, hi);
      acc[nt] = __builtin_amdgcn_wmma_f32_16x16x32_bf16(
          false, a, false, b0, (short)0, acc[nt], false, false);
      if (nt == 7) a = an;
      b0 = b1;
      b1 = b2;
    }
  }

  // C/D layout: element i -> (m = i + 8*hi, n = r)
#pragma unroll
  for (int nt = 0; nt < 8; ++nt) {
#pragma unroll
    for (int i = 0; i < 8; ++i) {
      Gi[(size_t)(m0 + i + 8 * hi) * G3H + (n0 + nt * 16 + r)] = acc[nt][i];
    }
  }
}

// ---------------------------------------------------------------------------
// Phase C: one GRU step. The workgroup's 16 h-rows (A operand, shared by all
// 8 waves) are staged in LDS once; each wave then computes the three gate
// tiles (r,z,n) with a flat (k-step, gate) loop, 2-deep B rotation, and A
// prefetched one gate ahead of each K boundary.
// Dynamic LDS: 16 * LDS_A_STRIDE = 16640 B.
// Grid: (B/16, H/128), block 256 (8 waves * 16 cols = 128 cols).
// ---------------------------------------------------------------------------
__global__ void gru_step_kernel(const unsigned short* __restrict__ hb_in,   // [B,H] bf16
                                unsigned short* __restrict__ hb_out,        // [B,H] bf16
                                float* __restrict__ hf,                     // [B,H] f32 (in/out)
                                const float* __restrict__ Gi_t,             // [B,3H] f32
                                const unsigned short* __restrict__ Whh,     // [3H,H] bf16
                                const float* __restrict__ b_ih,             // [3H]
                                const float* __restrict__ b_hh) {           // [3H]
  extern __shared__ char smem[];
  const int tid  = threadIdx.x;
  const int wave = tid >> 5;
  const int lane = tid & 31;
  const int r    = lane & 15;
  const int hi   = lane >> 4;
  const int m0   = blockIdx.x * 16;
  const int j0   = blockIdx.y * 128 + wave * 16;

  const bf16_t* H = (const bf16_t*)hb_in;
  const bf16_t* W = (const bf16_t*)Whh;
  const bf16_t* wp[3] = {
    W + (size_t)(0 * H_SZ + j0 + r) * H_SZ,
    W + (size_t)(1 * H_SZ + j0 + r) * H_SZ,
    W + (size_t)(2 * H_SZ + j0 + r) * H_SZ
  };
  __builtin_prefetch(wp[0], 0, 1);
  __builtin_prefetch(wp[1], 0, 1);
  __builtin_prefetch(wp[2], 0, 1);

  // Stage h rows m0..m0+15 (full K=512) into LDS: 16 threads/row, 64B each.
  {
    const int row = tid >> 4;
    const int seg = tid & 15;
    const uint4* src = (const uint4*)(H + (size_t)(m0 + row) * H_SZ + seg * 32);
    uint4* dst = (uint4*)(smem + (size_t)row * LDS_A_STRIDE + seg * 64);
    dst[0] = src[0]; dst[1] = src[1]; dst[2] = src[2]; dst[3] = src[3];
  }
  __syncthreads();

  v8f acc[3];
#pragma unroll
  for (int g = 0; g < 3; ++g) acc[g] = v8f_zero();

  // Flat pipeline: it = k-step*3 + gate, 48 iterations, 2-deep B rotation.
  v16bf a  = load_frag_lds(smem, LDS_A_STRIDE, r, 0, hi);
  v16bf an = a;
  v16bf b0 = load_frag(wp[0], 0, hi);
  v16bf b1 = load_frag(wp[1], 0, hi);
#pragma unroll
  for (int it = 0; it < 48; ++it) {
    const int g   = it % 3;
    const int it2 = it + 2;
    v16bf b2 = b1;
    if (it2 < 48)
      b2 = load_frag(wp[it2 % 3], (it2 / 3) * 32, hi);
    if (g == 1 && it2 < 48)   // next k-step's A fragment from LDS
      an = load_frag_lds(smem, LDS_A_STRIDE, r, (it2 / 3) * 32, hi);
    acc[g] = __builtin_amdgcn_wmma_f32_16x16x32_bf16(
        false, a, false, b0, (short)0, acc[g], false, false);
    if (g == 2) a = an;
    b0 = b1;
    b1 = b2;
  }

  const int j = j0 + r;
  const float br  = b_ih[j]             + b_hh[j];
  const float bz  = b_ih[H_SZ + j]      + b_hh[H_SZ + j];
  const float bin = b_ih[2 * H_SZ + j];
  const float bhn = b_hh[2 * H_SZ + j];

#pragma unroll
  for (int i = 0; i < 8; ++i) {
    const int row = m0 + i + 8 * hi;
    const float* gi = Gi_t + (size_t)row * G3H;
    float rg = sigmoidf_(acc[0][i] + gi[j] + br);
    float zg = sigmoidf_(acc[1][i] + gi[H_SZ + j] + bz);
    float ng = tanhf(gi[2 * H_SZ + j] + bin + rg * (acc[2][i] + bhn));
    size_t hidx = (size_t)row * H_SZ + j;
    float hold = hf[hidx];
    float hnew = (1.0f - zg) * ng + zg * hold;
    hf[hidx] = hnew;
    hb_out[hidx] = f32_to_bf16_bits(hnew);
  }
}

// ---------------------------------------------------------------------------
// Phase D1: out1 = tanh(h_final @ W1^T + b1), [256 x 512], K = 512.
// Grid: (B/16, M/128), block 256; 1 accumulator per wave. (Tiny; global-only.)
// ---------------------------------------------------------------------------
__global__ void head1_kernel(const unsigned short* __restrict__ hb,   // [B,H] bf16
                             const unsigned short* __restrict__ W1b,  // [M,H] bf16
                             const float* __restrict__ b1,            // [M]
                             float* __restrict__ out1) {              // [B,M] f32
  const int wave = threadIdx.x >> 5;
  const int lane = threadIdx.x & 31;
  const int r    = lane & 15;
  const int hi   = lane >> 4;
  const int m0   = blockIdx.x * 16;
  const int j0   = blockIdx.y * 128 + wave * 16;

  const bf16_t* H = (const bf16_t*)hb;
  const bf16_t* W = (const bf16_t*)W1b;
  const bf16_t* arow = H + (size_t)(m0 + r) * H_SZ;
  const bf16_t* brow = W + (size_t)(j0 + r) * H_SZ;

  v8f acc = v8f_zero();
  for (int k0 = 0; k0 < H_SZ; k0 += 32) {
    v16bf a = load_frag(arow, k0, hi);
    v16bf b = load_frag(brow, k0, hi);
    acc = __builtin_amdgcn_wmma_f32_16x16x32_bf16(
        false, a, false, b, (short)0, acc, false, false);
  }
  const int j = j0 + r;
  const float bias = b1[j];
#pragma unroll
  for (int i = 0; i < 8; ++i) {
    const int row = m0 + i + 8 * hi;
    out1[(size_t)row * M_SZ + j] = tanhf(acc[i] + bias);
  }
}

// ---------------------------------------------------------------------------
// Phase D2: value[b] = tanh(dot(out1[b,:], W2) + b2). Tiny.
// ---------------------------------------------------------------------------
__global__ void head2_kernel(const float* __restrict__ out1,
                             const float* __restrict__ W2,
                             const float* __restrict__ b2,
                             float* __restrict__ out) {
  int b = blockIdx.x * blockDim.x + threadIdx.x;
  if (b < B_SZ) {
    float s = 0.0f;
    for (int k = 0; k < M_SZ; ++k) s += out1[(size_t)b * M_SZ + k] * W2[k];
    out[b] = tanhf(s + b2[0]);
  }
}

// ---------------------------------------------------------------------------
// Host-side orchestration
// ---------------------------------------------------------------------------
extern "C" void kernel_launch(void* const* d_in, const int* in_sizes, int n_in,
                              void* d_out, int out_size, void* d_ws, size_t ws_size,
                              hipStream_t stream) {
  (void)in_sizes; (void)n_in; (void)out_size; (void)ws_size;

  const float* in_x   = (const float*)d_in[0];   // [S,B,D]
  const float* w_ih   = (const float*)d_in[1];   // [3H,D]
  const float* w_hh   = (const float*)d_in[2];   // [3H,H]
  const float* b_ih   = (const float*)d_in[3];   // [3H]
  const float* b_hh   = (const float*)d_in[4];   // [3H]
  const float* W1     = (const float*)d_in[5];   // [M,H]
  const float* b1     = (const float*)d_in[6];   // [M]
  const float* W2     = (const float*)d_in[7];   // [1,M]
  const float* b2     = (const float*)d_in[8];   // [1]
  float* out          = (float*)d_out;           // [B,1]

  // Workspace carve-up (256B aligned)
  char* ws = (char*)d_ws;
  size_t off = 0;
  auto carve = [&](size_t bytes) -> char* {
    char* p = ws + off;
    off += (bytes + 255) & ~(size_t)255;
    return p;
  };
  unsigned short* x_bf   = (unsigned short*)carve((size_t)SB * D_IN * 2);     // 67 MB
  unsigned short* wih_bf = (unsigned short*)carve((size_t)G3H * D_IN * 2);
  unsigned short* whh_bf = (unsigned short*)carve((size_t)G3H * H_SZ * 2);
  unsigned short* w1_bf  = (unsigned short*)carve((size_t)M_SZ * H_SZ * 2);
  float*          gi     = (float*)carve((size_t)SB * G3H * 4);               // 403 MB
  float*          h_f    = (float*)carve((size_t)B_SZ * H_SZ * 4);
  unsigned short* h_bf0  = (unsigned short*)carve((size_t)B_SZ * H_SZ * 2);
  unsigned short* h_bf1  = (unsigned short*)carve((size_t)B_SZ * H_SZ * 2);
  float*          out1   = (float*)carve((size_t)B_SZ * M_SZ * 4);

  // 1. Convert operands to bf16
  cvt_bf16_kernel<<<4096, 256, 0, stream>>>(in_x, x_bf, (size_t)SB * D_IN);
  cvt_bf16_kernel<<<512, 256, 0, stream>>>(w_ih, wih_bf, (size_t)G3H * D_IN);
  cvt_bf16_kernel<<<512, 256, 0, stream>>>(w_hh, whh_bf, (size_t)G3H * H_SZ);
  cvt_bf16_kernel<<<256, 256, 0, stream>>>(W1, w1_bf, (size_t)M_SZ * H_SZ);

  // 2. Zero initial hidden state (f32 + bf16 ping buffer)
  zero_kernel<<<128, 256, 0, stream>>>((unsigned int*)h_f, (size_t)B_SZ * H_SZ);
  zero_kernel<<<64, 256, 0, stream>>>((unsigned int*)h_bf0, (size_t)B_SZ * H_SZ / 2);

  // 3. Big parallel GEMM: Gi = x @ w_ih^T for all timesteps at once
  {
    dim3 grid(SB / 128, G3H / 128);
    gemm_gi_kernel<<<grid, 256, 128 * LDS_B_STRIDE, stream>>>(x_bf, wih_bf, gi);
  }

  // 4. Sequential GRU steps (stream order provides the dependency chain)
  {
    dim3 grid(B_SZ / 16, H_SZ / 128);
    for (int t = 0; t < S_LEN; ++t) {
      unsigned short* hin  = (t & 1) ? h_bf1 : h_bf0;
      unsigned short* hout = (t & 1) ? h_bf0 : h_bf1;
      const float* gi_t = gi + (size_t)t * B_SZ * G3H;
      gru_step_kernel<<<grid, 256, 16 * LDS_A_STRIDE, stream>>>(
          hin, hout, h_f, gi_t, whh_bf, b_ih, b_hh);
    }
  }

  // 5. Head: out1 = tanh(h @ W1^T + b1); value = tanh(out1 @ W2^T + b2)
  {
    unsigned short* h_final = (S_LEN & 1) ? h_bf0 : h_bf1;  // S_LEN=256 -> h_bf1
    dim3 grid(B_SZ / 16, M_SZ / 128);
    head1_kernel<<<grid, 256, 0, stream>>>(h_final, w1_bf, b1, out1);
    head2_kernel<<<1, 256, 0, stream>>>(out1, W2, b2, out);
  }
}
